// RNNModel_36163624632420
// MI455X (gfx1250) — compile-verified, matrix operand order checked by reading, not run on hardware
//
#include <hip/hip_runtime.h>
#include <hip/hip_bf16.h>
#include <stdint.h>

#define TT 128
#define BB 32
#define DD 1024
#define VV 32000
#define KK 8

typedef __bf16 bf16;
typedef __attribute__((ext_vector_type(16))) __bf16 v16bf;
typedef __attribute__((ext_vector_type(8)))  float  v8f;

__device__ __forceinline__ float sigmoidf_(float x) { return 1.0f / (1.0f + __expf(-x)); }

// CDNA5 async copy: 16B per lane, global -> LDS, no VGPR round trip (ASYNCcnt).
__device__ __forceinline__ void async_cp16(void* lds, const void* gaddr) {
    uint32_t l = (uint32_t)(uintptr_t)lds;            // low 32 bits = LDS address
    uint64_t g = (uint64_t)(uintptr_t)gaddr;
    asm volatile("global_load_async_to_lds_b128 %0, %1, off"
                 :: "v"(l), "v"(g) : "memory");
}
__device__ __forceinline__ void wait_async0() {
    asm volatile("s_wait_asynccnt 0x0" ::: "memory");
}

// ---------------------------------------------------------------------------
// Small helper kernels
// ---------------------------------------------------------------------------
__global__ void k_cvt_bf16(const float* __restrict__ src, bf16* __restrict__ dst, int n) {
    int i = blockIdx.x * 256 + threadIdx.x;
    if (i < n) dst[i] = (bf16)src[i];
}

__global__ void k_wcat(const float* __restrict__ whh, const float* __restrict__ whm,
                       bf16* __restrict__ Wcat) {
    int i = blockIdx.x * 256 + threadIdx.x;   // over D*D
    if (i < DD * DD) {
        int n = i >> 10, k = i & 1023;
        Wcat[(size_t)n * (2 * DD) + k]      = (bf16)whh[i];
        Wcat[(size_t)n * (2 * DD) + DD + k] = (bf16)whm[i];
    }
}

__global__ void k_addvec(const float* __restrict__ a, const float* __restrict__ b,
                         float* __restrict__ o, int n) {
    int i = blockIdx.x * 256 + threadIdx.x;
    if (i < n) o[i] = a[i] + b[i];
}

__global__ void k_init(const float* __restrict__ h0, const float* __restrict__ c0,
                       float* __restrict__ H, float* __restrict__ C,
                       float* __restrict__ Buf, bf16* __restrict__ HMbf) {
    int i = blockIdx.x * 256 + threadIdx.x;   // over K*B*D
    if (i < KK * BB * DD) {
        int k = i / (BB * DD);
        int r = i % (BB * DD);
        Buf[i] = (k == 0) ? h0[r] : 0.0f;
        if (k == 0) {
            H[r] = h0[r];
            C[r] = c0[r];
            int b = r / DD, d = r % DD;
            HMbf[(size_t)b * (2 * DD) + d] = (bf16)h0[r];
        }
    }
}

__global__ void k_embed(const int* __restrict__ tok, const float* __restrict__ encW,
                        bf16* __restrict__ emb) {
    int rowi = blockIdx.x;                     // 0 .. T*B-1
    int row  = tok[rowi];
    for (int j = threadIdx.x; j < DD; j += 256)
        emb[(size_t)rowi * DD + j] = (bf16)encW[(size_t)row * DD + j];
}

// ---------------------------------------------------------------------------
// Generic bf16 WMMA GEMM:  C[M,N] = A[M,K] * W[N,K]^T + bias
// BM in {32, 64}; BN = 128; BK = 32; 8 waves / block.
// Double-buffered LDS tiles filled by GLOBAL_LOAD_ASYNC_TO_LDS_B128 so the
// async DMA of chunk i+1 overlaps WMMA compute on chunk i.
// NT = non-temporal f32 stores (write-once outputs; keep L2 for weights).
// M multiple of BM, N multiple of 128, K multiple of 32 (guaranteed by caller).
// ---------------------------------------------------------------------------
template <int BM, bool OUTBF, bool NT>
__global__ void __launch_bounds__(256) k_gemm(
    const bf16* __restrict__ A, int lda,
    const bf16* __restrict__ W, int ldw,
    const float* __restrict__ bias,
    float* __restrict__ Cf, bf16* __restrict__ Cb, int ldc,
    int Kdim) {
    constexpr int BN = 128, BK = 32, LDT = 40;
    constexpr int WROWS = BM / 32;
    constexpr int WCOLS = 8 / WROWS;
    constexpr int WN = BN / WCOLS;
    constexpr int FN = WN / 16;

    __shared__ alignas(16) bf16 As[2][BM][LDT];
    __shared__ alignas(16) bf16 Ws[2][BN][LDT];

    const int tid  = threadIdx.x;
    const int lane = tid & 31;
    const int wid  = tid >> 5;
    const int wrow = (WROWS == 1) ? 0 : (wid & 1);
    const int wcol = (WROWS == 1) ? wid : (wid >> 1);
    const int row0 = blockIdx.y * BM;
    const int col0 = blockIdx.x * BN;

    const int kb = (lane < 16) ? 0 : 8;   // K base within fragment (ISA A/B layout)
    const int mr = lane & 15;             // row (A) / col (B) within tile
    const int mb = (lane < 16) ? 0 : 8;   // C-fragment M base

    // Fixed per-thread staging assignment (straight-line, no loops).
    const int ar = tid >> 2, as_ = tid & 3;                 // A seg (if tid < BM*4)
    const int wr0 = tid >> 2, ws0 = tid & 3;                // W seg 0
    const int wr1 = (tid + 256) >> 2, ws1 = (tid + 256) & 3;// W seg 1

    auto stage = [&](int p, int kc) {
        if (tid < BM * 4)
            async_cp16(&As[p][ar][as_ * 8],
                       &A[(size_t)(row0 + ar) * lda + kc + as_ * 8]);
        async_cp16(&Ws[p][wr0][ws0 * 8],
                   &W[(size_t)(col0 + wr0) * ldw + kc + ws0 * 8]);
        async_cp16(&Ws[p][wr1][ws1 * 8],
                   &W[(size_t)(col0 + wr1) * ldw + kc + ws1 * 8]);
    };

    const v8f zero8 = {0.f, 0.f, 0.f, 0.f, 0.f, 0.f, 0.f, 0.f};
    v8f c[2][FN];
#pragma unroll
    for (int fm = 0; fm < 2; ++fm)
#pragma unroll
        for (int fn = 0; fn < FN; ++fn) c[fm][fn] = zero8;

    const int nk = Kdim / BK;
    stage(0, 0);
    wait_async0();
    __syncthreads();

    for (int i = 0; i < nk; ++i) {
        const int p = i & 1;
        if (i + 1 < nk) stage(1 - p, (i + 1) * BK);   // overlap DMA with compute

        v16bf af[2];
#pragma unroll
        for (int fm = 0; fm < 2; ++fm) {
            uint32_t u[8];
#pragma unroll
            for (int v = 0; v < 8; ++v) {
                int k = kb + ((v < 4) ? 2 * v : 16 + 2 * (v - 4));
                u[v] = *reinterpret_cast<const uint32_t*>(
                    &As[p][wrow * 32 + fm * 16 + mr][k]);
            }
            __builtin_memcpy(&af[fm], u, 32);
        }
        v16bf bfr[FN];
#pragma unroll
        for (int fn = 0; fn < FN; ++fn) {
            uint32_t u[8];
#pragma unroll
            for (int v = 0; v < 8; ++v) {
                int k = kb + ((v < 4) ? 2 * v : 16 + 2 * (v - 4));
                u[v] = *reinterpret_cast<const uint32_t*>(
                    &Ws[p][wcol * WN + fn * 16 + mr][k]);
            }
            __builtin_memcpy(&bfr[fn], u, 32);
        }
#pragma unroll
        for (int fm = 0; fm < 2; ++fm)
#pragma unroll
            for (int fn = 0; fn < FN; ++fn)
                c[fm][fn] = __builtin_amdgcn_wmma_f32_16x16x32_bf16(
                    false, af[fm], false, bfr[fn], (short)0, c[fm][fn], false, false);

        wait_async0();       // chunk i+1 fully landed in LDS
        __syncthreads();     // all waves done reading buffer p
    }

#pragma unroll
    for (int fm = 0; fm < 2; ++fm)
#pragma unroll
        for (int fn = 0; fn < FN; ++fn) {
            int cidx = col0 + wcol * WN + fn * 16 + mr;
            float bb = bias ? bias[cidx] : 0.0f;
#pragma unroll
            for (int v = 0; v < 8; ++v) {
                int r = row0 + wrow * 32 + fm * 16 + mb + v;
                float val = c[fm][fn][v] + bb;
                if (OUTBF)      Cb[(size_t)r * ldc + cidx] = (bf16)val;
                else if (NT)    __builtin_nontemporal_store(val, &Cf[(size_t)r * ldc + cidx]);
                else            Cf[(size_t)r * ldc + cidx] = val;
            }
        }
}

// ---------------------------------------------------------------------------
// Fused per-step kernel: gates = Xproj[t] + h @ W_hh^T  -> LSTM update -> h,c
// Each block owns a 64-wide d-slice; 8 waves cover the 4 gates x 64 columns.
// Double-buffered async staging; the gate-regroup buffer reuses the tile LDS
// (union) since it is only needed after the K loop.
// ---------------------------------------------------------------------------
__global__ void __launch_bounds__(256) k_gates_lstm(
    const bf16* __restrict__ Hbf,   // [32][2048] (h half used, lda = 2048)
    const bf16* __restrict__ Whh,   // [4096][1024] bf16
    const float* __restrict__ Xp,   // Xproj + t*B*4D, [32][4096], biases included
    float* __restrict__ Cst, float* __restrict__ Hst,
    bf16* __restrict__ HMbf) {
    union ShU {
        struct { bf16 As[2][32][40]; bf16 Ws[2][256][40]; } t;
        float gbuf[4][32][64];
    };
    __shared__ alignas(16) ShU sh;

    const int tid  = threadIdx.x;
    const int lane = tid & 31;
    const int wid  = tid >> 5;
    const int d0   = blockIdx.x * 64;
    const int gi   = wid >> 1;      // gate index i,f,g,o
    const int half = wid & 1;       // which 32-col half of the 64-slice
    const int kb = (lane < 16) ? 0 : 8;
    const int mr = lane & 15;
    const int mb = (lane < 16) ? 0 : 8;

    auto stage = [&](int p, int kc) {
        if (tid < 32 * 4) {
            int r = tid >> 2, s = tid & 3;
            async_cp16(&sh.t.As[p][r][s * 8], &Hbf[(size_t)r * (2 * DD) + kc + s * 8]);
        }
#pragma unroll
        for (int i = 0; i < 4; ++i) {
            int e = tid + i * 256;
            int r = e >> 2, s = e & 3;
            int w = r >> 5, loc = r & 31;
            size_t grow = (size_t)(w >> 1) * DD + d0 + (w & 1) * 32 + loc;
            async_cp16(&sh.t.Ws[p][r][s * 8], &Whh[grow * DD + kc + s * 8]);
        }
    };

    const v8f zero8 = {0.f, 0.f, 0.f, 0.f, 0.f, 0.f, 0.f, 0.f};
    v8f c[2][2];
#pragma unroll
    for (int fm = 0; fm < 2; ++fm)
#pragma unroll
        for (int fn = 0; fn < 2; ++fn) c[fm][fn] = zero8;

    constexpr int NK = DD / 32;
    stage(0, 0);
    wait_async0();
    __syncthreads();

    for (int i = 0; i < NK; ++i) {
        const int p = i & 1;
        if (i + 1 < NK) stage(1 - p, (i + 1) * 32);   // overlap DMA with compute

        v16bf af[2], bfr[2];
#pragma unroll
        for (int fm = 0; fm < 2; ++fm) {
            uint32_t u[8];
#pragma unroll
            for (int v = 0; v < 8; ++v) {
                int k = kb + ((v < 4) ? 2 * v : 16 + 2 * (v - 4));
                u[v] = *reinterpret_cast<const uint32_t*>(&sh.t.As[p][fm * 16 + mr][k]);
            }
            __builtin_memcpy(&af[fm], u, 32);
        }
#pragma unroll
        for (int fn = 0; fn < 2; ++fn) {
            uint32_t u[8];
#pragma unroll
            for (int v = 0; v < 8; ++v) {
                int k = kb + ((v < 4) ? 2 * v : 16 + 2 * (v - 4));
                u[v] = *reinterpret_cast<const uint32_t*>(
                    &sh.t.Ws[p][wid * 32 + fn * 16 + mr][k]);
            }
            __builtin_memcpy(&bfr[fn], u, 32);
        }
#pragma unroll
        for (int fm = 0; fm < 2; ++fm)
#pragma unroll
            for (int fn = 0; fn < 2; ++fn)
                c[fm][fn] = __builtin_amdgcn_wmma_f32_16x16x32_bf16(
                    false, af[fm], false, bfr[fn], (short)0, c[fm][fn], false, false);

        wait_async0();
        __syncthreads();
    }

    // Tiles fully consumed (final barrier above) -> reuse LDS as gate buffer.
    // Scatter gate pre-activations, add input-projection (+biases).
#pragma unroll
    for (int fm = 0; fm < 2; ++fm)
#pragma unroll
        for (int fn = 0; fn < 2; ++fn) {
            int dloc = half * 32 + fn * 16 + mr;
            int gcol = gi * DD + d0 + dloc;
#pragma unroll
            for (int v = 0; v < 8; ++v) {
                int b = fm * 16 + mb + v;
                sh.gbuf[gi][b][dloc] = c[fm][fn][v] + Xp[(size_t)b * (4 * DD) + gcol];
            }
        }
    __syncthreads();

    // Elementwise LSTM cell update for this d-slice.
    for (int e = tid; e < 32 * 64; e += 256) {
        int b = e >> 6, dl = e & 63, d = d0 + dl;
        float ig = sh.gbuf[0][b][dl];
        float fg = sh.gbuf[1][b][dl];
        float gg = sh.gbuf[2][b][dl];
        float og = sh.gbuf[3][b][dl];
        float co = Cst[(size_t)b * DD + d];
        float cn = sigmoidf_(fg) * co + sigmoidf_(ig) * tanhf(gg);
        float hn = sigmoidf_(og) * tanhf(cn);
        Cst[(size_t)b * DD + d] = cn;
        Hst[(size_t)b * DD + d] = hn;
        HMbf[(size_t)b * (2 * DD) + d] = (bf16)hn;   // h half of [h | m]
    }
}

// ---------------------------------------------------------------------------
// Windowed attention over the K=8 hidden buffer + circular-buffer append.
// Single workgroup (B=32, K=8 is tiny).
// ---------------------------------------------------------------------------
__global__ void __launch_bounds__(256) k_attn(
    const float* __restrict__ H, float* __restrict__ Buf,
    const float* __restrict__ wmi, const float* __restrict__ wmib,
    const float* __restrict__ wmh, const float* __restrict__ wmhb,
    bf16* __restrict__ HMbf, int t) {
    __shared__ float ei[KK][BB];
    __shared__ float att[KK][BB];
    __shared__ float et[BB];
    const int tid = threadIdx.x;

    {   // e_i[k][b] = buf[k][b] . wmi
        int k = tid >> 5, b = tid & 31;
        float acc = 0.f;
        const float* br = &Buf[((size_t)k * BB + b) * DD];
        for (int d = 0; d < DD; ++d) acc += br[d] * wmi[d];
        ei[k][b] = acc + wmib[0];
    }
    if (tid < BB) {   // e_t[b] = h[b] . wmh
        float acc = 0.f;
        const float* hr = &H[(size_t)tid * DD];
        for (int d = 0; d < DD; ++d) acc += hr[d] * wmh[d];
        et[tid] = acc + wmhb[0];
    }
    __syncthreads();

    if (tid < BB) {   // masked softmax over k
        float sc[KK];
        float mx = -1e30f;
#pragma unroll
        for (int s = 0; s < KK; ++s) {
            int md = (t - s) % KK; if (md < 0) md += KK;
            int entry = t - md;
            float v = (entry >= 0) ? (ei[s][tid] + et[tid]) : -1e9f;
            sc[s] = v; mx = fmaxf(mx, v);
        }
        float sum = 0.f;
#pragma unroll
        for (int s = 0; s < KK; ++s) { float e_ = __expf(sc[s] - mx); sc[s] = e_; sum += e_; }
        float inv = 1.0f / sum;
#pragma unroll
        for (int s = 0; s < KK; ++s) att[s][tid] = sc[s] * inv;
    }
    __syncthreads();

    // m[b][d] -> m half of [h | m] (bf16)
    for (int e = tid; e < BB * DD; e += 256) {
        int b = e >> 10, d = e & 1023;
        float acc = 0.f;
#pragma unroll
        for (int k = 0; k < KK; ++k)
            acc += att[k][b] * Buf[((size_t)k * BB + b) * DD + d];
        HMbf[(size_t)b * (2 * DD) + DD + d] = (bf16)acc;
    }
    __syncthreads();

    // append h_t at slot (t+1) % K (after all buf reads)
    int slot = (t + 1) & (KK - 1);
    for (int e = tid; e < BB * DD; e += 256) {
        int b = e >> 10, d = e & 1023;
        Buf[((size_t)slot * BB + b) * DD + d] = H[(size_t)b * DD + d];
    }
}

// ---------------------------------------------------------------------------
// Host orchestration
// ---------------------------------------------------------------------------
extern "C" void kernel_launch(void* const* d_in, const int* in_sizes, int n_in,
                              void* d_out, int out_size, void* d_ws, size_t ws_size,
                              hipStream_t stream) {
    (void)in_sizes; (void)n_in; (void)out_size; (void)ws_size;
    const int*   tokens = (const int*)d_in[0];
    const float* h0     = (const float*)d_in[1];
    const float* c0     = (const float*)d_in[2];
    const float* enc_W  = (const float*)d_in[3];
    const float* dec_b  = (const float*)d_in[4];
    const float* W_ih   = (const float*)d_in[5];
    const float* W_hh   = (const float*)d_in[6];
    const float* b_ih   = (const float*)d_in[7];
    const float* b_hh   = (const float*)d_in[8];
    const float* wmi_w  = (const float*)d_in[9];
    const float* wmi_b  = (const float*)d_in[10];
    const float* wmh_w  = (const float*)d_in[11];
    const float* wmh_b  = (const float*)d_in[12];
    const float* whh_w  = (const float*)d_in[13];
    const float* whh_b  = (const float*)d_in[14];
    const float* whm_w  = (const float*)d_in[15];
    const float* whm_b  = (const float*)d_in[16];
    float* out = (float*)d_out;

    char* wp = (char*)d_ws;
    auto alloc = [&](size_t bytes) -> char* {
        char* p = wp;
        wp += (bytes + 255) & ~(size_t)255;
        return p;
    };
    bf16*  encW_bf = (bf16*)alloc((size_t)VV * DD * 2);
    bf16*  Wih_bf  = (bf16*)alloc((size_t)4 * DD * DD * 2);
    bf16*  Whh_bf  = (bf16*)alloc((size_t)4 * DD * DD * 2);
    bf16*  Wcat_bf = (bf16*)alloc((size_t)DD * 2 * DD * 2);
    bf16*  emb_bf  = (bf16*)alloc((size_t)TT * BB * DD * 2);
    bf16*  raw_bf  = (bf16*)alloc((size_t)TT * BB * DD * 2);
    bf16*  hm_bf   = (bf16*)alloc((size_t)BB * 2 * DD * 2);
    float* Xproj   = (float*)alloc((size_t)TT * BB * 4 * DD * 4);
    float* Hst     = (float*)alloc((size_t)BB * DD * 4);
    float* Cst     = (float*)alloc((size_t)BB * DD * 4);
    float* Buf     = (float*)alloc((size_t)KK * BB * DD * 4);
    float* bias_g  = (float*)alloc((size_t)4 * DD * 4);
    float* bias_o  = (float*)alloc((size_t)DD * 4);

    // Prologue: bf16 weight conversions, fused biases, state init, embedding.
    { int n = VV * DD;     k_cvt_bf16<<<(n + 255) / 256, 256, 0, stream>>>(enc_W, encW_bf, n); }
    { int n = 4 * DD * DD; k_cvt_bf16<<<(n + 255) / 256, 256, 0, stream>>>(W_ih, Wih_bf, n); }
    { int n = 4 * DD * DD; k_cvt_bf16<<<(n + 255) / 256, 256, 0, stream>>>(W_hh, Whh_bf, n); }
    { int n = DD * DD;     k_wcat<<<(n + 255) / 256, 256, 0, stream>>>(whh_w, whm_w, Wcat_bf); }
    k_addvec<<<(4 * DD + 255) / 256, 256, 0, stream>>>(b_ih, b_hh, bias_g, 4 * DD);
    k_addvec<<<(DD + 255) / 256, 256, 0, stream>>>(whh_b, whm_b, bias_o, DD);
    { int n = KK * BB * DD; k_init<<<(n + 255) / 256, 256, 0, stream>>>(h0, c0, Hst, Cst, Buf, hm_bf); }
    k_embed<<<TT * BB, 256, 0, stream>>>(tokens, enc_W, emb_bf);

    // Hoisted input projection for all timesteps (biases folded in):
    // Xproj[T*B, 4D] = emb @ W_ih^T + (b_ih + b_hh)
    k_gemm<64, false, false><<<dim3(4 * DD / 128, TT * BB / 64), 256, 0, stream>>>(
        emb_bf, DD, Wih_bf, DD, bias_g, Xproj, nullptr, 4 * DD, DD);

    // Sequential recurrence: 3 launches per step.
    for (int t = 0; t < TT; ++t) {
        k_gates_lstm<<<DD / 64, 256, 0, stream>>>(
            hm_bf, Whh_bf, Xproj + (size_t)t * BB * 4 * DD, Cst, Hst, hm_bf);
        k_attn<<<1, 256, 0, stream>>>(Hst, Buf, wmi_w, wmi_b, wmh_w, wmh_b, hm_bf, t);
        // raw[t] = [h | m] @ [whh ; whm]^T + (whh_b + whm_b), K = 2048
        k_gemm<32, true, false><<<dim3(DD / 128, 1), 256, 0, stream>>>(
            hm_bf, 2 * DD, Wcat_bf, 2 * DD, bias_o,
            nullptr, raw_bf + (size_t)t * BB * DD, DD, 2 * DD);
    }

    // Decoder: logits[T*B, V] = raw @ enc_W^T + dec_b   (dominant GEMM, 268 GF)
    // NT stores: 524 MB write-once output must not evict enc_W from L2.
    k_gemm<64, false, true><<<dim3(VV / 128, TT * BB / 64), 256, 0, stream>>>(
        raw_bf, DD, encW_bf, DD, dec_b, out, nullptr, VV, DD);
}
